// QuaternionLSTM_6004364280306
// MI455X (gfx1250) — matrix operand order repaired and checked by reading, not compiled, not run on hardware
//
#include <hip/hip_runtime.h>

typedef __attribute__((ext_vector_type(16))) __bf16 v16bf;
typedef __attribute__((ext_vector_type(8)))  float  v8f;

#define T_STEPS 1024
#define HID     128
#define NB      32
#define NL      2
#define NBLK    32
// packed weights per layer: [128 mt][32 kt][32 lane][16 elem] bf16 (kt<16: Wx, kt>=16: Wh)
#define WPK_PER_LAYER (128*32*32*16)

// ---------------------------------------------------------------------------
// Pack the structured quaternion weight matrix into bf16 WMMA A-fragment order.
// Row m = g*512 + h*4 + s ; col k = i*4 + q ; coef = sgn[s][q]*W[g][h][i][pidx[s][q]]
// ---------------------------------------------------------------------------
__global__ __launch_bounds__(256) void qlstm_pack(const float* __restrict__ Wx,
                                                  const float* __restrict__ Wh,
                                                  __bf16* __restrict__ wpk) {
    const int   pidx[4][4] = {{0,1,2,3},{1,0,3,2},{2,3,0,1},{3,2,1,0}};
    const float sgnt[4][4] = {{ 1.f,-1.f,-1.f,-1.f},
                              { 1.f, 1.f,-1.f, 1.f},
                              { 1.f, 1.f, 1.f,-1.f},
                              { 1.f,-1.f, 1.f, 1.f}};
    long idx = (long)blockIdx.x * 256 + threadIdx.x;   // [l][mt][kt][lane][e]
    int e    = (int)(idx)        & 15;
    int lane = (int)(idx >> 4)   & 31;
    int kt   = (int)(idx >> 9)   & 31;
    int mt   = (int)(idx >> 14)  & 127;
    int l    = (int)(idx >> 21);

    int m    = mt * 16 + (lane & 15);
    int half = lane >> 4;
    int kl   = e + 16 * half;
    int k    = (kt & 15) * 32 + kl;

    const float* W = (kt >= 16) ? Wh : Wx;
    int g = m >> 9, h = (m >> 2) & 127, s = m & 3;
    int i = k >> 2, q = k & 3;
    float v = sgnt[s][q] * W[(((long)(l*4 + g) * 128 + h) * 128 + i) * 4 + pidx[s][q]];
    wpk[idx] = (__bf16)v;
}

// zero the bf16 activation buffer and the grid-barrier counter (every call)
__global__ __launch_bounds__(256) void qlstm_init(__bf16* __restrict__ act,
                                                  unsigned* __restrict__ barcnt) {
    int i = blockIdx.x * 256 + threadIdx.x;           // 65536 threads
    if (i < NL * NB * 1024) act[i] = (__bf16)0.0f;
    if (i == 0) *barcnt = 0u;
}

// ---------------------------------------------------------------------------
// Monotonic arrival-count grid barrier (counter zeroed by qlstm_init per call).
// ---------------------------------------------------------------------------
__device__ __forceinline__ void grid_barrier(unsigned* cnt, unsigned target) {
    __threadfence();
    __syncthreads();
    if (threadIdx.x == 0) {
        __hip_atomic_fetch_add(cnt, 1u, __ATOMIC_ACQ_REL, __HIP_MEMORY_SCOPE_AGENT);
        while (__hip_atomic_load(cnt, __ATOMIC_ACQUIRE, __HIP_MEMORY_SCOPE_AGENT) < target)
            __builtin_amdgcn_s_sleep(1);
    }
    __syncthreads();
    __threadfence();
}

__device__ __forceinline__ float sigf(float x) { return 1.0f / (1.0f + __expf(-x)); }

__device__ __forceinline__ float4 ham4(float4 a, float4 b) {
    float4 r;
    r.x = a.x*b.x - a.y*b.y - a.z*b.z - a.w*b.w;
    r.y = a.x*b.y + a.y*b.x + a.z*b.w - a.w*b.z;
    r.z = a.x*b.z - a.y*b.w + a.z*b.x + a.w*b.y;
    r.w = a.x*b.w + a.y*b.z - a.z*b.y + a.w*b.x;
    return r;
}

__device__ __forceinline__ float4 ln4(float4 v, const float* g, const float* bt) {
    float m  = 0.25f * (v.x + v.y + v.z + v.w);
    float dx = v.x - m, dy = v.y - m, dz = v.z - m, dw = v.w - m;
    float var = 0.25f * (dx*dx + dy*dy + dz*dz + dw*dw);
    float rs = rsqrtf(var + 1e-5f);
    float4 r;
    r.x = dx*rs*g[0] + bt[0]; r.y = dy*rs*g[1] + bt[1];
    r.z = dz*rs*g[2] + bt[2]; r.w = dw*rs*g[3] + bt[3];
    return r;
}

// ---------------------------------------------------------------------------
// Persistent kernel: 32 blocks x 256 threads. Block bid owns m-tiles
// {bid, 32+bid, 64+bid, 96+bid} = all 4 gates for hidden quats [4*bid..4*bid+3].
// Per phase (t,l): stage act in LDS -> 8 wave-tiles of WMMA (wave = (gate, nt))
// -> z tiles to LDS -> in-block cell update (c state in registers) -> grid barrier.
// ---------------------------------------------------------------------------
__global__ __launch_bounds__(256) void qlstm_persistent(
        const float* __restrict__ x,   const __bf16* __restrict__ wpk,
        const float* __restrict__ bx,  const float* __restrict__ bh,
        const float* __restrict__ lg,  const float* __restrict__ lb,
        __bf16* __restrict__ act, float* __restrict__ yout,
        float* __restrict__ hn, float* __restrict__ cn,
        unsigned* __restrict__ barcnt) {
    __shared__ __align__(32) __bf16 sact[NB * 1024];   // 64 KB
    float* zbuf = (float*)sact;                        // 8 KB alias: [g][row16][col32]

    const int tid = threadIdx.x, bid = blockIdx.x;
    const int wid = tid >> 5, lane = tid & 31;
    const int g   = wid >> 1, nt = wid & 1;            // wave tile: gate, batch half
    const int ncol = lane & 15, halfk = (lane >> 4);

    // elem-thread identity (tid < 128): b = tid>>2, local quat hl = tid&3
    const int eb = tid >> 2, hl = tid & 3;
    const int hq = bid * 4 + hl;
    float4 creg[NL] = {make_float4(0,0,0,0), make_float4(0,0,0,0)};

    // hoist (bx+bh) gate biases: fixed (g,hq,s) per elem thread for all steps
    float bsum[NL][4][4];
    if (tid < 128) {
        #pragma unroll
        for (int l = 0; l < NL; ++l)
            #pragma unroll
            for (int gg = 0; gg < 4; ++gg)
                #pragma unroll
                for (int s = 0; s < 4; ++s) {
                    int bi = l * 2048 + (gg * 128 + hq) * 4 + s;
                    bsum[l][gg][s] = bx[bi] + bh[bi];
                }
    }

    unsigned bar = 0;
    for (int t = 0; t < T_STEPS; ++t) {
        #pragma unroll
        for (int l = 0; l < NL; ++l) {
            const __bf16* actl = act + l * NB * 1024;
            // ---- stage activations [32][1024] bf16 into LDS ----
            if (l == 0) {
                const float* xt = x + (size_t)t * 512;
                for (int c = tid; c < 4096; c += 256) {       // x_t f32 -> bf16
                    int b = c >> 7, f4 = (c & 127) * 4;
                    float4 v = *(const float4*)(xt + (size_t)b * (T_STEPS * 512) + f4);
                    __bf16* d = sact + b * 1024 + f4;
                    d[0] = (__bf16)v.x; d[1] = (__bf16)v.y;
                    d[2] = (__bf16)v.z; d[3] = (__bf16)v.w;
                }
                for (int c = tid; c < 2048; c += 256) {       // h0_prev bf16 copy
                    int b = c >> 6, o = (c & 63) * 8;
                    *(uint4*)(sact + b * 1024 + 512 + o) =
                        *(const uint4*)(actl + b * 1024 + 512 + o);
                }
            } else {
                for (int c = tid; c < 4096; c += 256) {       // [h0_new ; h1_prev]
                    int b = c >> 7, o = (c & 127) * 8;
                    *(uint4*)(sact + b * 1024 + o) =
                        *(const uint4*)(actl + b * 1024 + o);
                }
            }
            __syncthreads();

            // ---- GEMM: wave tile mt = g*32 + bid, software-pipelined loads ----
            const __bf16* abase = wpk + (size_t)l * WPK_PER_LAYER
                                + (size_t)(g * 32 + bid) * (32 * 32 * 16) + lane * 16;
            const __bf16* bbase = sact + (nt * 16 + ncol) * 1024 + halfk * 16;
            v16bf a  = *(const v16bf*)(abase);
            v16bf bm = *(const v16bf*)(bbase);
            v8f acc = {};
            #pragma unroll 8
            for (int kc = 0; kc < 32; ++kc) {
                int kn = (kc + 1) & 31;
                v16bf an = *(const v16bf*)(abase + (size_t)kn * (32 * 16));
                v16bf bn = *(const v16bf*)(bbase + kn * 32);
                if ((kc & 7) == 0)
                    __builtin_prefetch((const void*)(abase + (size_t)((kc + 8) & 31) * (32 * 16)), 0, 3);
                acc = __builtin_amdgcn_wmma_f32_16x16x32_bf16(false, a, false, bm,
                                                              (short)0, acc, false, false);
                a = an; bm = bn;
            }
            __syncthreads();   // all waves done reading sact (zbuf aliases it)

            // ---- z tiles to LDS: row = r + 8*halfk, col = nt*16 + ncol ----
            {
                int row = 8 * halfk, col = nt * 16 + ncol;
                #pragma unroll
                for (int r = 0; r < 8; ++r)
                    zbuf[(g * 16 + row + r) * 32 + col] = acc[r];
            }
            __syncthreads();

            // ---- in-block cell update (threads 0..127) ----
            if (tid < 128) {
                float zg[4][4];
                #pragma unroll
                for (int gg = 0; gg < 4; ++gg)
                    #pragma unroll
                    for (int s = 0; s < 4; ++s)
                        zg[gg][s] = zbuf[(gg * 16 + hl * 4 + s) * 32 + eb] + bsum[l][gg][s];

                float4 gi = make_float4(sigf(zg[0][0]),  sigf(zg[0][1]),  sigf(zg[0][2]),  sigf(zg[0][3]));
                float4 gf = make_float4(sigf(zg[1][0]),  sigf(zg[1][1]),  sigf(zg[1][2]),  sigf(zg[1][3]));
                float4 gc = make_float4(tanhf(zg[2][0]), tanhf(zg[2][1]), tanhf(zg[2][2]), tanhf(zg[2][3]));
                float4 go = make_float4(sigf(zg[3][0]),  sigf(zg[3][1]),  sigf(zg[3][2]),  sigf(zg[3][3]));

                const float* lgl = lg + l * 8;
                const float* lbl = lb + l * 8;
                float4 a1 = ham4(gf, creg[l]), a2 = ham4(gi, gc);
                float4 craw = make_float4(a1.x + a2.x, a1.y + a2.y, a1.z + a2.z, a1.w + a2.w);
                float4 cnew = ln4(craw, lgl, lbl);
                creg[l] = cnew;

                float4 tc   = make_float4(tanhf(cnew.x), tanhf(cnew.y), tanhf(cnew.z), tanhf(cnew.w));
                float4 hnew = ln4(ham4(go, tc), lgl + 4, lbl + 4);

                int ao = eb * 1024 + hq * 4;
                __bf16* ah = act + l * NB * 1024 + 512 + ao;   // this layer h_prev slot
                ah[0] = (__bf16)hnew.x; ah[1] = (__bf16)hnew.y;
                ah[2] = (__bf16)hnew.z; ah[3] = (__bf16)hnew.w;
                if (l == 0) {
                    __bf16* an2 = act + NB * 1024 + ao;        // layer-1 input slot
                    an2[0] = (__bf16)hnew.x; an2[1] = (__bf16)hnew.y;
                    an2[2] = (__bf16)hnew.z; an2[3] = (__bf16)hnew.w;
                } else {
                    *(float4*)(yout + ((size_t)(eb * T_STEPS + t) * HID + hq) * 4) = hnew;
                }
                if (t == T_STEPS - 1) {
                    *(float4*)(hn + ((size_t)(l * NB + eb) * HID + hq) * 4) = hnew;
                    *(float4*)(cn + ((size_t)(l * NB + eb) * HID + hq) * 4) = cnew;
                }
            }
            // ---- publish h, sync whole grid ----
            bar += NBLK;
            grid_barrier(barcnt, bar);
        }
    }
}

// ---------------------------------------------------------------------------
extern "C" void kernel_launch(void* const* d_in, const int* in_sizes, int n_in,
                              void* d_out, int out_size, void* d_ws, size_t ws_size,
                              hipStream_t stream) {
    (void)in_sizes; (void)n_in; (void)out_size; (void)ws_size;
    const float* x  = (const float*)d_in[0];
    const float* Wx = (const float*)d_in[1];
    const float* Wh = (const float*)d_in[2];
    const float* bx = (const float*)d_in[3];
    const float* bh = (const float*)d_in[4];
    const float* lg = (const float*)d_in[5];
    const float* lb = (const float*)d_in[6];

    float* out  = (float*)d_out;
    float* yout = out;
    float* hn   = out + (size_t)NB * T_STEPS * HID * 4;   // (L,B,H,4)
    float* cn   = hn + (size_t)NL * NB * HID * 4;         // (L,B,H,4)

    char* wsb      = (char*)d_ws;
    __bf16*  wpk   = (__bf16*)wsb;                                     // 8 MB
    __bf16*  act   = (__bf16*)(wsb + (size_t)2 * WPK_PER_LAYER * 2);   // 128 KB
    unsigned* bcnt = (unsigned*)(wsb + (size_t)2 * WPK_PER_LAYER * 2
                                     + (size_t)NL * NB * 1024 * 2);

    qlstm_pack<<<(2 * WPK_PER_LAYER) / 256, 256, 0, stream>>>(Wx, Wh, wpk);
    qlstm_init<<<(NL * NB * 1024) / 256, 256, 0, stream>>>(act, bcnt);
    qlstm_persistent<<<NBLK, 256, 0, stream>>>(x, wpk, bx, bh, lg, lb,
                                               act, yout, hn, cn, bcnt);
}
